// RegressEncoder_33200097198273
// MI455X (gfx1250) — matrix operand order repaired and checked by reading, not compiled, not run on hardware
//
#include <hip/hip_runtime.h>

typedef __attribute__((ext_vector_type(16))) _Float16 v16h;
typedef __attribute__((ext_vector_type(8)))  _Float16 v8h;
typedef __attribute__((ext_vector_type(8)))  float    v8f;

#define NEG_SLOPE 0.2f
constexpr int BB = 4, HH = 128, WW = 128;

__device__ __forceinline__ float lrelu(float v){ return v >= 0.f ? v : NEG_SLOPE * v; }

// ---------------- layout prep kernels ----------------

// (B,C,H,W) f32 -> (B,H,W,C) f16
__global__ void k_nchw_to_nhwc_f16(const float* __restrict__ in, _Float16* __restrict__ out,
                                   int C) {
  size_t tot = (size_t)BB * HH * WW * C;
  for (size_t i = (size_t)blockIdx.x * blockDim.x + threadIdx.x; i < tot;
       i += (size_t)gridDim.x * blockDim.x) {
    int c = (int)(i % C); size_t t = i / C;
    int x = (int)(t % WW); t /= WW;
    int y = (int)(t % HH); int b = (int)(t / HH);
    out[i] = (_Float16)in[(((size_t)b * C + c) * HH + y) * WW + x];
  }
}

// w: (O,Ci,3,3) f32 -> wp f16 packed [tap][cinBlock][n(OPad)][kpair(16)][2]
// => one lane's B-fragment (16 f16, kp = half*8..half*8+7) is 32 contiguous bytes.
__global__ void k_prep_w(const float* __restrict__ w, _Float16* __restrict__ wp,
                         int O, int Ci, int OPad) {
  int nCb = Ci >> 5;
  size_t tot = (size_t)9 * nCb * OPad * 16 * 2;
  for (size_t i = (size_t)blockIdx.x * blockDim.x + threadIdx.x; i < tot;
       i += (size_t)gridDim.x * blockDim.x) {
    int e  = (int)(i & 1); size_t t = i >> 1;
    int kp = (int)(t % 16);   t /= 16;
    int o  = (int)(t % OPad); t /= OPad;
    int cb = (int)(t % nCb);  int tap = (int)(t / nCb);
    int c  = cb * 32 + ((kp & 7) >> 2) * 16 + (kp >> 3) * 8 + (kp & 3) * 2 + e;
    float v = 0.f;
    if (o < O) v = w[(((size_t)o * Ci + c) * 3 + tap / 3) * 3 + (tap % 3)];
    wp[i] = (_Float16)v;
  }
}

// ---------------- fragment helpers ----------------

// A 16x32 f16 fragment: lane half h, pixel row ml; VGPR j<4: K=h*8+2j(+1); j>=4: K=16+h*8+2(j-4)
__device__ __forceinline__ v16h load_afrag(const _Float16* pa /* row base + hlf*8 */) {
  v8h lo = *(const v8h*)(pa);
  v8h hi = *(const v8h*)(pa + 16);
  return __builtin_shufflevector(lo, hi, 0,1,2,3,4,5,6,7,8,9,10,11,12,13,14,15);
}

// ---------------- implicit-GEMM 3x3 conv (stride1, SAME) via WMMA ----------------
// in: f16 NHWC. wp packed per k_prep_w. MODE 0: LeakyReLU->f16 NHWC; MODE 1: linear->f32 NHWC.
template<int Cin, int CoutPad, int CoutReal, int MODE>
__global__ __launch_bounds__(256) void k_conv3x3(
    const _Float16* __restrict__ in, const _Float16* __restrict__ wp,
    const float* __restrict__ bias, void* __restrict__ out) {
  constexpr int nCb = Cin / 32;
  constexpr int NW  = CoutPad / 16;
  constexpr int T   = NW * 32;
  __shared__ _Float16 patch[3 * 18 * Cin];
  const int x0 = blockIdx.x * 16, y = blockIdx.y, b = blockIdx.z;
  const int lane = threadIdx.x, wv = threadIdx.y;
  const int tid = wv * 32 + lane;

  // cooperative stage of the 3x18xCin input patch (zero-padded borders)
  constexpr int tot = 3 * 18 * Cin;
  for (int i = tid; i < tot; i += T) {
    int c = i % Cin; int t = i / Cin;
    int cc = t % 18; int r = t / 18;
    int gy = y + r - 1, gx = x0 + cc - 1;
    _Float16 v = (_Float16)0.f;
    if (gy >= 0 && gy < HH && gx >= 0 && gx < WW)
      v = in[(((size_t)b * HH + gy) * WW + gx) * Cin + c];
    patch[(r * 18 + cc) * Cin + c] = v;
  }
  __syncthreads();

  const int hlf = lane >> 4, ml = lane & 15;
  const int n = wv * 16 + ml;                 // output channel (B/D column)
  float bv = (n < CoutReal) ? bias[n] : 0.f;
  v8f acc;
  #pragma unroll
  for (int j = 0; j < 8; ++j) acc[j] = bv;

  #pragma unroll
  for (int tap = 0; tap < 9; ++tap) {
    const _Float16* prow = &patch[((tap / 3) * 18 + (ml + tap % 3)) * Cin + hlf * 8];
    #pragma unroll
    for (int cb = 0; cb < nCb; ++cb) {
      v16h a  = load_afrag(prow + cb * 32);
      v16h bf = *(const v16h*)(wp + (((size_t)(tap * nCb + cb) * CoutPad + n) * 16 + hlf * 8) * 2);
      acc = __builtin_amdgcn_wmma_f32_16x16x32_f16(false, a, false, bf,
                                                   (short)0, acc, false, false);
    }
  }

  if (MODE == 0) {
    _Float16* o = (_Float16*)out;
    #pragma unroll
    for (int j = 0; j < 8; ++j) {
      int m = j + (hlf << 3);
      if (n < CoutReal)
        o[(((size_t)b * HH + y) * WW + (x0 + m)) * CoutReal + n] = (_Float16)lrelu(acc[j]);
    }
  } else {
    float* o = (float*)out;
    #pragma unroll
    for (int j = 0; j < 8; ++j) {
      int m = j + (hlf << 3);
      if (n < CoutReal)
        o[(((size_t)b * HH + y) * WW + (x0 + m)) * CoutReal + n] = acc[j];
    }
  }
}

// ---------------- deformable 3x3 conv: bilinear gather into LDS + WMMA ----------------
// h: f16 NHWC (C ch); off: f32 NHWC (18 ch, dy/dx per tap).
// MODE 0: LeakyReLU->f16 NHWC. MODE 1: LeakyReLU->f32 NCHW (final output).
template<int C, int CoutPad, int CoutReal, int MODE>
__global__ __launch_bounds__(256) void k_deform(
    const _Float16* __restrict__ h, const float* __restrict__ off,
    const _Float16* __restrict__ wp, const float* __restrict__ bias,
    void* __restrict__ out) {
  constexpr int nCb = C / 32;
  constexpr int NW  = CoutPad / 16;
  constexpr int T   = NW * 32;
  __shared__ _Float16 sampA[2][16 * 32];     // double-buffered 16px x 32ch tile
  const int x0 = blockIdx.x * 16, y = blockIdx.y, b = blockIdx.z;
  const int lane = threadIdx.x, wv = threadIdx.y;
  const int tid = wv * 32 + lane;
  const int hlf = lane >> 4, ml = lane & 15;
  const int n = wv * 16 + ml;
  float bv = (n < CoutReal) ? bias[n] : 0.f;
  v8f acc;
  #pragma unroll
  for (int j = 0; j < 8; ++j) acc[j] = bv;

  int it = 0;
  #pragma unroll
  for (int tap = 0; tap < 9; ++tap) {
    const int tr = tap / 3 - 1, tc = tap % 3 - 1;
    #pragma unroll
    for (int cb = 0; cb < nCb; ++cb) {
      _Float16* sb = sampA[it & 1];
      // cooperative bilinear sampling (uniform trip count: 512 % T == 0)
      #pragma unroll
      for (int i0 = 0; i0 < 512; i0 += T) {
        int i = i0 + tid;
        int p = i >> 5, c = i & 31;
        int gx = x0 + p;
        const float* op = off + (((size_t)b * HH + y) * WW + gx) * 18 + tap * 2;
        float py = (float)(y + tr) + op[0];
        float px = (float)(gx + tc) + op[1];
        float fy = floorf(py), fx = floorf(px);
        int iy = (int)fy, ixx = (int)fx;
        float wy = py - fy, wx = px - fx;
        int ch = cb * 32 + c;
        float v = 0.f;
        #pragma unroll
        for (int cy = 0; cy < 2; ++cy) {
          int yy = iy + cy;
          if (yy < 0 || yy >= HH) continue;
          float wwy = cy ? wy : (1.f - wy);
          #pragma unroll
          for (int cx = 0; cx < 2; ++cx) {
            int xx = ixx + cx;
            if (xx < 0 || xx >= WW) continue;
            float ww = wwy * (cx ? wx : (1.f - wx));
            v += ww * (float)h[(((size_t)b * HH + yy) * WW + xx) * C + ch];
          }
        }
        sb[p * 32 + c] = (_Float16)v;
      }
      __syncthreads();   // write(b) visible; ping-pong makes next write safe w/o 2nd barrier

      v16h a  = load_afrag(sb + ml * 32 + hlf * 8);
      v16h bf = *(const v16h*)(wp + (((size_t)(tap * nCb + cb) * CoutPad + n) * 16 + hlf * 8) * 2);
      acc = __builtin_amdgcn_wmma_f32_16x16x32_f16(false, a, false, bf,
                                                   (short)0, acc, false, false);
      ++it;
    }
  }

  if (MODE == 0) {
    _Float16* o = (_Float16*)out;
    #pragma unroll
    for (int j = 0; j < 8; ++j) {
      int m = j + (hlf << 3);
      if (n < CoutReal)
        o[(((size_t)b * HH + y) * WW + (x0 + m)) * CoutReal + n] = (_Float16)lrelu(acc[j]);
    }
  } else {
    float* o = (float*)out;                  // NCHW final output
    #pragma unroll
    for (int j = 0; j < 8; ++j) {
      int m = j + (hlf << 3);
      if (n < CoutReal)
        o[(((size_t)b * CoutReal + n) * HH + y) * WW + (x0 + m)] = lrelu(acc[j]);
    }
  }
}

// ---------------- host orchestration ----------------

extern "C" void kernel_launch(void* const* d_in, const int* in_sizes, int n_in,
                              void* d_out, int out_size, void* d_ws, size_t ws_size,
                              hipStream_t stream) {
  (void)in_sizes; (void)n_in; (void)out_size; (void)ws_size;
  const float* x = (const float*)d_in[0];
  struct StP { const float *cw, *cb, *ow, *ob, *dw, *db; };
  StP st[3];
  for (int s = 0; s < 3; ++s) {
    st[s].cw = (const float*)d_in[1 + 6 * s];
    st[s].cb = (const float*)d_in[2 + 6 * s];
    st[s].ow = (const float*)d_in[3 + 6 * s];
    st[s].ob = (const float*)d_in[4 + 6 * s];
    st[s].dw = (const float*)d_in[5 + 6 * s];
    st[s].db = (const float*)d_in[6 + 6 * s];
  }

  char* ws = (char*)d_ws;
  size_t oofs = 0;
  auto carve = [&](size_t bytes) -> char* {
    char* p = ws + oofs; oofs += (bytes + 255) & ~(size_t)255; return p;
  };
  _Float16* bufX   = (_Float16*)carve((size_t)BB * HH * WW * 64 * 2);   //  8 MB
  _Float16* bufH   = (_Float16*)carve((size_t)BB * HH * WW * 128 * 2);  // 16 MB
  _Float16* bufG0  = (_Float16*)carve((size_t)BB * HH * WW * 128 * 2);  // 16 MB
  _Float16* bufG1  = (_Float16*)carve((size_t)BB * HH * WW * 128 * 2);  // 16 MB
  float*    bufOff = (float*)   carve((size_t)BB * HH * WW * 18 * 4);   // 4.5 MB
  _Float16* wpC    = (_Float16*)carve((size_t)9 * 4 * 128 * 16 * 2 * 2);
  _Float16* wpO    = (_Float16*)carve((size_t)9 * 4 * 32 * 16 * 2 * 2);
  _Float16* wpD    = (_Float16*)carve((size_t)9 * 4 * 128 * 16 * 2 * 2);

  k_nchw_to_nhwc_f16<<<dim3(2048), dim3(256), 0, stream>>>(x, bufX, 64);

  dim3 grd(WW / 16, HH, BB);

  // ---- stage 1: Cin=64 -> Co=128 ----
  k_prep_w<<<dim3(128), dim3(256), 0, stream>>>(st[0].cw, wpC, 128, 64, 128);
  k_prep_w<<<dim3(64),  dim3(256), 0, stream>>>(st[0].ow, wpO, 18, 128, 32);
  k_prep_w<<<dim3(128), dim3(256), 0, stream>>>(st[0].dw, wpD, 128, 128, 128);
  k_conv3x3<64, 128, 128, 0><<<grd, dim3(32, 8), 0, stream>>>(bufX, wpC, st[0].cb, bufH);
  k_conv3x3<128, 32, 18, 1><<<grd, dim3(32, 2), 0, stream>>>(bufH, wpO, st[0].ob, bufOff);
  k_deform<128, 128, 128, 0><<<grd, dim3(32, 8), 0, stream>>>(bufH, bufOff, wpD, st[0].db, bufG0);

  // ---- stage 2: Cin=128 -> Co=128 ----
  k_prep_w<<<dim3(128), dim3(256), 0, stream>>>(st[1].cw, wpC, 128, 128, 128);
  k_prep_w<<<dim3(64),  dim3(256), 0, stream>>>(st[1].ow, wpO, 18, 128, 32);
  k_prep_w<<<dim3(128), dim3(256), 0, stream>>>(st[1].dw, wpD, 128, 128, 128);
  k_conv3x3<128, 128, 128, 0><<<grd, dim3(32, 8), 0, stream>>>(bufG0, wpC, st[1].cb, bufH);
  k_conv3x3<128, 32, 18, 1><<<grd, dim3(32, 2), 0, stream>>>(bufH, wpO, st[1].ob, bufOff);
  k_deform<128, 128, 128, 0><<<grd, dim3(32, 8), 0, stream>>>(bufH, bufOff, wpD, st[1].db, bufG1);

  // ---- stage 3: Cin=128 -> Co=64, final f32 NCHW to d_out ----
  k_prep_w<<<dim3(128), dim3(256), 0, stream>>>(st[2].cw, wpC, 64, 128, 64);
  k_prep_w<<<dim3(64),  dim3(256), 0, stream>>>(st[2].ow, wpO, 18, 64, 32);
  k_prep_w<<<dim3(128), dim3(256), 0, stream>>>(st[2].dw, wpD, 64, 64, 64);
  k_conv3x3<128, 64, 64, 0><<<grd, dim3(32, 4), 0, stream>>>(bufG1, wpC, st[2].cb, bufH);
  k_conv3x3<64, 32, 18, 1><<<grd, dim3(32, 2), 0, stream>>>(bufH, wpO, st[2].ob, bufOff);
  k_deform<64, 64, 64, 1><<<grd, dim3(32, 4), 0, stream>>>(bufH, bufOff, wpD, st[2].db, d_out);
}
